// CustomAttnProcessor_60498909331719
// MI455X (gfx1250) — compile-verified, matrix operand order checked by reading, not run on hardware
//
#include <hip/hip_runtime.h>
#include <hip/hip_bf16.h>
#include <math.h>

// ---------------------------------------------------------------------------
// Types for CDNA5 (gfx1250) WMMA: wave32, V_WMMA_F32_16X16X32_BF16
// ---------------------------------------------------------------------------
typedef __bf16 bf16_t;
typedef __bf16 v16bf __attribute__((ext_vector_type(16)));
typedef __bf16 v8bf  __attribute__((ext_vector_type(8)));
typedef float  v8f   __attribute__((ext_vector_type(8)));

union FragBF { v16bf v; v8bf h[2]; };

// ---------------------------------------------------------------------------
// Tensor Data Mover: 2-D tile DMA global -> LDS (CDNA5 ISA)
// D# group0: count=1 | lds_addr | global_addr[56:0] | type=2
// D# group1: data_size=2B, tensor_dim0/1, tile_dim0/1, tensor_dim0_stride
// Issued once per workgroup by wave 0; completion via s_wait_tensorcnt.
// ---------------------------------------------------------------------------
#ifndef USE_TDM
#define USE_TDM 1
#endif
#if defined(__HIP_DEVICE_COMPILE__) && USE_TDM
#define TDM_ENABLED 1
#else
#define TDM_ENABLED 0
#endif

#if TDM_ENABLED
typedef unsigned int v4u __attribute__((ext_vector_type(4)));
typedef unsigned int v8u __attribute__((ext_vector_type(8)));

__device__ __forceinline__ void tdm_load_tile_2d(unsigned ldsOff, const void* gptr,
                                                 unsigned tileD0, unsigned tileD1,
                                                 unsigned tensorD0, unsigned tensorD1,
                                                 unsigned strideElems) {
  unsigned long long ga = (unsigned long long)gptr;
  v4u g0;
  g0[0] = (unsigned)__builtin_amdgcn_readfirstlane(1u);            // count=1
  g0[1] = (unsigned)__builtin_amdgcn_readfirstlane(ldsOff);        // lds_addr (bytes)
  g0[2] = (unsigned)__builtin_amdgcn_readfirstlane((unsigned)ga);  // global_addr[31:0]
  g0[3] = (unsigned)__builtin_amdgcn_readfirstlane(
      (unsigned)((ga >> 32) & 0x01ffffffu) | 0x80000000u);         // ga[56:32] | type=2
  v8u g1;
  g1[0] = (unsigned)__builtin_amdgcn_readfirstlane(0x00010000u);   // data_size=1 (2 bytes)
  g1[1] = (unsigned)__builtin_amdgcn_readfirstlane((tensorD0 & 0xffffu) << 16);
  g1[2] = (unsigned)__builtin_amdgcn_readfirstlane((tensorD0 >> 16) |
                                                   ((tensorD1 & 0xffffu) << 16));
  g1[3] = (unsigned)__builtin_amdgcn_readfirstlane((tensorD1 >> 16) |
                                                   ((tileD0 & 0xffffu) << 16));
  g1[4] = (unsigned)__builtin_amdgcn_readfirstlane(tileD1 & 0xffffu);
  g1[5] = (unsigned)__builtin_amdgcn_readfirstlane(strideElems);   // dim0 stride (elems)
  g1[6] = (unsigned)__builtin_amdgcn_readfirstlane(0u);
  g1[7] = (unsigned)__builtin_amdgcn_readfirstlane(0u);
  asm volatile("tensor_load_to_lds %0, %1" :: "s"(g0), "s"(g1) : "memory");
}
#endif

// ---------------------------------------------------------------------------
// fp32 -> bf16 conversion (grid-stride)
// ---------------------------------------------------------------------------
__global__ void cvt_bf16_kernel(const float* __restrict__ s, bf16_t* __restrict__ d, long n) {
  long i  = (long)blockIdx.x * blockDim.x + threadIdx.x;
  long st = (long)gridDim.x * blockDim.x;
  for (; i < n; i += st) d[i] = (bf16_t)s[i];
}

// Copy hidden_states (B,1024,D) into x (B,1054,D) visual slots
__global__ void scatter_visual_kernel(const float* __restrict__ src, float* __restrict__ dst,
                                      int rowsPerB, int dstRowsPerB, int D, int Bn) {
  long total = (long)Bn * rowsPerB * D;
  long i  = (long)blockIdx.x * blockDim.x + threadIdx.x;
  long st = (long)gridDim.x * blockDim.x;
  long per = (long)rowsPerB * D;
  for (; i < total; i += st) {
    long b = i / per, r = i % per;
    dst[b * (long)dstRowsPerB * D + r] = src[i];
  }
}

// out = a + tanh(alpha) * b   (in-place allowed: out == a)
__global__ void residual_kernel(float* __restrict__ out, const float* __restrict__ a,
                                const float* __restrict__ b, const float* __restrict__ alpha,
                                long n) {
  float t = tanhf(alpha[0]);
  long i  = (long)blockIdx.x * blockDim.x + threadIdx.x;
  long st = (long)gridDim.x * blockDim.x;
  for (; i < n; i += st) out[i] = a[i] + t * b[i];
}

// ---------------------------------------------------------------------------
// LayerNorm (one 256-thread block per row), f32 in -> bf16 out
// ---------------------------------------------------------------------------
__global__ __launch_bounds__(256)
void layernorm_bf16_kernel(const float* __restrict__ x, const float* __restrict__ g,
                           const float* __restrict__ bta, bf16_t* __restrict__ out, int D) {
  __shared__ float sm[8], sv[8];
  long row = blockIdx.x;
  const float* xr = x + row * (long)D;
  float s = 0.f, s2 = 0.f;
  for (int i = threadIdx.x; i < D; i += 256) { float v = xr[i]; s += v; s2 += v * v; }
#pragma unroll
  for (int m = 16; m; m >>= 1) { s += __shfl_xor(s, m, 32); s2 += __shfl_xor(s2, m, 32); }
  int wid = threadIdx.x >> 5;
  if ((threadIdx.x & 31) == 0) { sm[wid] = s; sv[wid] = s2; }
  __syncthreads();
  float ts = 0.f, t2 = 0.f;
#pragma unroll
  for (int i = 0; i < 8; ++i) { ts += sm[i]; t2 += sv[i]; }
  float mean = ts / (float)D;
  float var  = t2 / (float)D - mean * mean;
  float rstd = rsqrtf(var + 1e-5f);
  bf16_t* orow = out + row * (long)D;
  for (int i = threadIdx.x; i < D; i += 256)
    orow[i] = (bf16_t)((xr[i] - mean) * rstd * g[i] + bta[i]);
}

// ---------------------------------------------------------------------------
// bf16 WMMA GEMM: C = A(MxK) @ B0(KxN, row-major ldb) [+bias0]
// Block tile 128x128, K-step 32, 8 waves (2x4), wave tile 64x32 -> 8 WMMA/step.
// A tile staged by one TENSOR_LOAD_TO_LDS per K-step (zero-fill handles M edge).
// GEGLU: also acc2 = A @ B1 (+bias1); out = (acc+bias0) * gelu_erf(acc2+bias1).
// Output f32 (Cf) or bf16 (Cb). grid = (N/128, ceil(M/128), batch).
// ---------------------------------------------------------------------------
template <bool GEGLU, bool OUTBF16>
__global__ __launch_bounds__(256)
void gemm_bf16_kernel(const bf16_t* __restrict__ A, long aBatch, int M, int lda,
                      const bf16_t* __restrict__ B0, const bf16_t* __restrict__ B1, int ldb,
                      const float* __restrict__ bias0, const float* __restrict__ bias1,
                      float* __restrict__ Cf, bf16_t* __restrict__ Cb, long cBatch, int ldc,
                      int N, int K) {
  __shared__ bf16_t As[128][32];                       // [m][k]
  __shared__ bf16_t Bs[128][32];                       // transposed: [n][k]
  __shared__ bf16_t Bs2[GEGLU ? 128 : 1][32];          // gate weights, transposed

  const int tid  = threadIdx.x;
  const int wid  = tid >> 5, lane = tid & 31;
  const int llo  = lane & 15, lhi = lane >> 4;
  const int wm   = wid >> 2, wn = wid & 3;             // 2 x 4 wave grid
  const int m0   = blockIdx.y * 128, n0 = blockIdx.x * 128;
  const long zA  = (long)blockIdx.z * aBatch;
  const long zC  = (long)blockIdx.z * cBatch;

  v8f acc[4][2];
  v8f acc2[GEGLU ? 4 : 1][GEGLU ? 2 : 1];
#pragma unroll
  for (int mt = 0; mt < 4; ++mt)
#pragma unroll
    for (int nt = 0; nt < 2; ++nt)
#pragma unroll
      for (int r = 0; r < 8; ++r) acc[mt][nt][r] = 0.f;
  if (GEGLU) {
#pragma unroll
    for (int mt = 0; mt < 4; ++mt)
#pragma unroll
      for (int nt = 0; nt < 2; ++nt)
#pragma unroll
        for (int r = 0; r < 8; ++r) acc2[mt][nt][r] = 0.f;
  }

  for (int k0 = 0; k0 < K; k0 += 32) {
    __syncthreads();
    // ---- stage A tile (128 rows x 32 k): Tensor Data Mover DMA ----
#if TDM_ENABLED
    if (wid == 0) {
      tdm_load_tile_2d((unsigned)(unsigned long long)(const void*)&As[0][0],
                       A + zA + (long)m0 * lda + k0,
                       /*tileD0=*/32u, /*tileD1=*/128u,
                       /*tensorD0=*/(unsigned)(K - k0), /*tensorD1=*/(unsigned)(M - m0),
                       /*stride=*/(unsigned)lda);
    }
#else
#pragma unroll
    for (int it = 0; it < 2; ++it) {
      int idx = tid + it * 256;
      int row = idx >> 2, kq = (idx & 3) * 8;
      int gr  = m0 + row;
      v8bf val;
#pragma unroll
      for (int i = 0; i < 8; ++i) val[i] = (bf16_t)0.f;
      if (gr < M) val = *(const v8bf*)(A + zA + (long)gr * lda + k0 + kq);
      *(v8bf*)&As[row][kq] = val;
    }
#endif
    // ---- stage B tile(s) (32 k x 128 n), transposed into LDS ----
#pragma unroll
    for (int it = 0; it < 2; ++it) {
      int idx = tid + it * 256;
      int kr = idx >> 4, nc = (idx & 15) * 8;
      v8bf val = *(const v8bf*)(B0 + (long)(k0 + kr) * ldb + n0 + nc);
#pragma unroll
      for (int j = 0; j < 8; ++j) Bs[nc + j][kr] = val[j];
      if (GEGLU) {
        v8bf v2 = *(const v8bf*)(B1 + (long)(k0 + kr) * ldb + n0 + nc);
#pragma unroll
        for (int j = 0; j < 8; ++j) Bs2[nc + j][kr] = v2[j];
      }
    }
    // ---- prefetch next B K-tile (global_prefetch_b8) ----
    if (k0 + 32 < K)
      __builtin_prefetch(B0 + (long)(k0 + 32 + (tid >> 4)) * ldb + n0 + (tid & 15) * 8, 0, 1);
#if TDM_ENABLED
    if (wid == 0) __builtin_amdgcn_s_wait_tensorcnt(0);
#endif
    __syncthreads();

    // ---- fragments ----
    v16bf af[4], bfr[2], bfr2[2];
#pragma unroll
    for (int mt = 0; mt < 4; ++mt) {
      const bf16_t* ap = &As[wm * 64 + mt * 16 + llo][lhi * 8];
      FragBF f; f.h[0] = *(const v8bf*)ap; f.h[1] = *(const v8bf*)(ap + 16);
      af[mt] = f.v;
    }
#pragma unroll
    for (int nt = 0; nt < 2; ++nt) {
      const bf16_t* bp = &Bs[wn * 32 + nt * 16 + llo][lhi * 8];
      FragBF f; f.h[0] = *(const v8bf*)bp; f.h[1] = *(const v8bf*)(bp + 16);
      bfr[nt] = f.v;
      if (GEGLU) {
        const bf16_t* b2 = &Bs2[wn * 32 + nt * 16 + llo][lhi * 8];
        FragBF f2; f2.h[0] = *(const v8bf*)b2; f2.h[1] = *(const v8bf*)(b2 + 16);
        bfr2[nt] = f2.v;
      }
    }
    // ---- WMMA ----
#pragma unroll
    for (int mt = 0; mt < 4; ++mt)
#pragma unroll
      for (int nt = 0; nt < 2; ++nt) {
        acc[mt][nt] = __builtin_amdgcn_wmma_f32_16x16x32_bf16(
            false, af[mt], false, bfr[nt], (short)0, acc[mt][nt], false, false);
        if (GEGLU)
          acc2[mt][nt] = __builtin_amdgcn_wmma_f32_16x16x32_bf16(
              false, af[mt], false, bfr2[nt], (short)0, acc2[mt][nt], false, false);
      }
  }

  // ---- epilogue ----
#pragma unroll
  for (int mt = 0; mt < 4; ++mt)
#pragma unroll
    for (int nt = 0; nt < 2; ++nt)
#pragma unroll
      for (int r = 0; r < 8; ++r) {
        int row = m0 + wm * 64 + mt * 16 + r + 8 * lhi;
        int col = n0 + wn * 32 + nt * 16 + llo;
        if (row < M) {
          float vv = acc[mt][nt][r] + (bias0 ? bias0[col] : 0.f);
          if (GEGLU) {
            float gg = acc2[mt][nt][r] + (bias1 ? bias1[col] : 0.f);
            float ge = 0.5f * gg * (1.f + erff(gg * 0.70710678118654752f));
            vv = vv * ge;
          }
          long ci = zC + (long)row * ldc + col;
          if (OUTBF16) Cb[ci] = (bf16_t)vv; else Cf[ci] = vv;
        }
      }
}

// ---------------------------------------------------------------------------
// Flash attention (online softmax), bf16 in/out, head_dim = 64.
// grid = (Nq/128, H, B); 8 waves, each owns 16 query rows.
// K tile staged by TDM (zero-fill clips trailing keys); V staged manually
// (needs transpose). sim via 4 WMMA, P via LDS, P@V via 4 WMMA.
// Optional int32 mask (>0 keeps, else -1e30), matching the reference.
// ---------------------------------------------------------------------------
__global__ __launch_bounds__(256)
void flash_attn_kernel(const bf16_t* __restrict__ q, const bf16_t* __restrict__ k,
                       const bf16_t* __restrict__ v, const int* __restrict__ mask,
                       bf16_t* __restrict__ out,
                       int Nq, int Nkv, int qStride, int kStride, int oStride,
                       long qBatch, long kBatch, long oBatch,
                       long maskBatch, long maskHead, int maskRow, float scale) {
  __shared__ bf16_t Ks[32][64];    // [key][c]
  __shared__ bf16_t Vt[64][32];    // transposed: [c][key]
  __shared__ bf16_t Ps[8][16][32]; // per-wave P staging

  const int tid  = threadIdx.x;
  const int wid  = tid >> 5, lane = tid & 31;
  const int llo  = lane & 15, lhi = lane >> 4;
  const int h    = blockIdx.y, b = blockIdx.z;
  const int qRow0 = blockIdx.x * 128 + wid * 16;

  // Q fragments, head dim 64 = 2 K-chunks of 32
  v16bf qa[2];
#pragma unroll
  for (int ch = 0; ch < 2; ++ch) {
    int row = qRow0 + llo;
    const bf16_t* qp = q + (long)b * qBatch + (long)row * qStride + h * 64 + ch * 32 + lhi * 8;
    FragBF f; f.h[0] = *(const v8bf*)qp; f.h[1] = *(const v8bf*)(qp + 16);
    qa[ch] = f.v;
  }

  float mrun[8], lrun[8];
  v8f oacc[4];
#pragma unroll
  for (int r = 0; r < 8; ++r) { mrun[r] = -1e30f; lrun[r] = 0.f; }
#pragma unroll
  for (int nt2 = 0; nt2 < 4; ++nt2)
#pragma unroll
    for (int r = 0; r < 8; ++r) oacc[nt2][r] = 0.f;

  for (int j0 = 0; j0 < Nkv; j0 += 32) {
    __syncthreads();
#if TDM_ENABLED
    if (wid == 0) {
      tdm_load_tile_2d((unsigned)(unsigned long long)(const void*)&Ks[0][0],
                       k + (long)b * kBatch + (long)j0 * kStride + h * 64,
                       /*tileD0=*/64u, /*tileD1=*/32u,
                       /*tensorD0=*/64u, /*tensorD1=*/(unsigned)(Nkv - j0),
                       /*stride=*/(unsigned)kStride);
    }
    {
      int kr = tid >> 3, cc = (tid & 7) * 8;
      int gk = j0 + kr;
      v8bf vv;
#pragma unroll
      for (int i = 0; i < 8; ++i) vv[i] = (bf16_t)0.f;
      if (gk < Nkv)
        vv = *(const v8bf*)(v + (long)b * kBatch + (long)gk * kStride + h * 64 + cc);
#pragma unroll
      for (int i = 0; i < 8; ++i) Vt[cc + i][kr] = vv[i];
    }
    if (wid == 0) __builtin_amdgcn_s_wait_tensorcnt(0);
#else
    {
      int kr = tid >> 3, cc = (tid & 7) * 8;
      int gk = j0 + kr;
      v8bf kv, vv;
#pragma unroll
      for (int i = 0; i < 8; ++i) { kv[i] = (bf16_t)0.f; vv[i] = (bf16_t)0.f; }
      if (gk < Nkv) {
        kv = *(const v8bf*)(k + (long)b * kBatch + (long)gk * kStride + h * 64 + cc);
        vv = *(const v8bf*)(v + (long)b * kBatch + (long)gk * kStride + h * 64 + cc);
      }
      *(v8bf*)&Ks[kr][cc] = kv;
#pragma unroll
      for (int i = 0; i < 8; ++i) Vt[cc + i][kr] = vv[i];
    }
#endif
    __syncthreads();

    // sim = Q @ K^T  (two 16-key tiles)
    v8f s[2];
#pragma unroll
    for (int nt = 0; nt < 2; ++nt) {
#pragma unroll
      for (int i = 0; i < 8; ++i) s[nt][i] = 0.f;
#pragma unroll
      for (int ch = 0; ch < 2; ++ch) {
        const bf16_t* bp = &Ks[nt * 16 + llo][ch * 32 + lhi * 8];
        FragBF f; f.h[0] = *(const v8bf*)bp; f.h[1] = *(const v8bf*)(bp + 16);
        s[nt] = __builtin_amdgcn_wmma_f32_16x16x32_bf16(
            false, qa[ch], false, f.v, (short)0, s[nt], false, false);
      }
    }

    // scale + mask
#pragma unroll
    for (int nt = 0; nt < 2; ++nt)
#pragma unroll
      for (int r = 0; r < 8; ++r) {
        int ii = qRow0 + r + 8 * lhi;
        int jj = j0 + nt * 16 + llo;
        float sv = s[nt][r] * scale;
        if (jj >= Nkv) sv = -__builtin_inff();
        else if (mask) {
          int mval = mask[(long)b * maskBatch + (long)h * maskHead + (long)ii * maskRow + jj];
          if (mval <= 0) sv = -1e30f;
        }
        s[nt][r] = sv;
      }

    // online softmax per row, stage P into LDS (A-fragment layout source)
#pragma unroll
    for (int r = 0; r < 8; ++r) {
      float ml = fmaxf(s[0][r], s[1][r]);
#pragma unroll
      for (int m = 8; m; m >>= 1) ml = fmaxf(ml, __shfl_xor(ml, m, 32));
      float newm = fmaxf(mrun[r], ml);
      float corr = __expf(mrun[r] - newm);
      float p0 = __expf(s[0][r] - newm);
      float p1 = __expf(s[1][r] - newm);
      float sum = p0 + p1;
#pragma unroll
      for (int m = 8; m; m >>= 1) sum += __shfl_xor(sum, m, 32);
      lrun[r] = lrun[r] * corr + sum;
      mrun[r] = newm;
#pragma unroll
      for (int nt2 = 0; nt2 < 4; ++nt2) oacc[nt2][r] *= corr;
      Ps[wid][r + 8 * lhi][llo]      = (bf16_t)p0;
      Ps[wid][r + 8 * lhi][16 + llo] = (bf16_t)p1;
    }
    __syncthreads();

    // P fragment + P @ V
    v16bf pa;
    {
      const bf16_t* pp = &Ps[wid][llo][lhi * 8];
      FragBF f; f.h[0] = *(const v8bf*)pp; f.h[1] = *(const v8bf*)(pp + 16);
      pa = f.v;
    }
#pragma unroll
    for (int nt2 = 0; nt2 < 4; ++nt2) {
      const bf16_t* vp = &Vt[nt2 * 16 + llo][lhi * 8];
      FragBF f; f.h[0] = *(const v8bf*)vp; f.h[1] = *(const v8bf*)(vp + 16);
      oacc[nt2] = __builtin_amdgcn_wmma_f32_16x16x32_bf16(
          false, pa, false, f.v, (short)0, oacc[nt2], false, false);
    }
  }

  // normalize + write
#pragma unroll
  for (int nt2 = 0; nt2 < 4; ++nt2)
#pragma unroll
    for (int r = 0; r < 8; ++r) {
      int ii = qRow0 + r + 8 * lhi;
      int cc = nt2 * 16 + llo;
      if (ii < Nq) {
        float ov = oacc[nt2][r] / lrun[r];
        out[(long)b * oBatch + (long)ii * oStride + h * 64 + cc] = (bf16_t)ov;
      }
    }
}

// ---------------------------------------------------------------------------
// Orchestration
// ---------------------------------------------------------------------------
extern "C" void kernel_launch(void* const* d_in, const int* in_sizes, int n_in,
                              void* d_out, int out_size, void* d_ws, size_t ws_size,
                              hipStream_t stream) {
  (void)in_sizes; (void)n_in; (void)out_size; (void)ws_size;
  const int Bn = 8, Nv = 1024, Lobj = 30, Nx = 1054, Ltxt = 77;
  const int D = 1280, Dt = 768, IC = 512, FF = 5120;

  // ---- workspace allocator ----
  char* wsb = (char*)d_ws;
  size_t off = 0;
  auto alloc = [&](size_t bytes) -> void* {
    off = (off + 255) & ~(size_t)255;
    void* p = wsb + off;
    off += bytes;
    return p;
  };
  bf16_t* wLinear = (bf16_t*)alloc((size_t)Dt * D * 2);
  bf16_t* wGeglu  = (bf16_t*)alloc((size_t)D * 2 * FF * 2);
  bf16_t* wFfout  = (bf16_t*)alloc((size_t)FF * D * 2);
  bf16_t* wQ      = (bf16_t*)alloc((size_t)D * IC * 2);
  bf16_t* wK      = (bf16_t*)alloc((size_t)D * IC * 2);
  bf16_t* wV      = (bf16_t*)alloc((size_t)D * IC * 2);
  bf16_t* wCo     = (bf16_t*)alloc((size_t)IC * D * 2);
  bf16_t* wAq     = (bf16_t*)alloc((size_t)D * D * 2);
  bf16_t* wAk     = (bf16_t*)alloc((size_t)Dt * D * 2);
  bf16_t* wAv     = (bf16_t*)alloc((size_t)Dt * D * 2);
  bf16_t* wAo     = (bf16_t*)alloc((size_t)D * D * 2);
  bf16_t* encB    = (bf16_t*)alloc((size_t)Bn * Ltxt * Dt * 2);
  bf16_t* objB    = (bf16_t*)alloc((size_t)Bn * Lobj * Dt * 2);
  float*  xF      = (float*)alloc((size_t)Bn * Nx * D * 4);   // also attnproj / ffout f32
  bf16_t* lnB     = (bf16_t*)alloc((size_t)Bn * Nx * D * 2);  // ln1 / ln2 / ln3
  bf16_t* qB      = (bf16_t*)alloc((size_t)Bn * Nx * IC * 2);
  bf16_t* kB      = (bf16_t*)alloc((size_t)Bn * Nx * IC * 2);
  bf16_t* vB      = (bf16_t*)alloc((size_t)Bn * Nx * IC * 2);
  bf16_t* ctxSelf = (bf16_t*)alloc((size_t)Bn * Nv * IC * 2);
  float*  hs1     = (float*)alloc((size_t)Bn * Nv * D * 4);
  bf16_t* ffin    = (bf16_t*)alloc((size_t)Bn * Nv * FF * 2); // also qx/kx/vx/ctx2
  bf16_t* qxB  = ffin;
  bf16_t* kxB  = ffin + (size_t)Bn * Nv * D;
  bf16_t* vxB  = kxB + (size_t)Bn * Ltxt * D;
  bf16_t* ctx2 = vxB + (size_t)Bn * Ltxt * D;

  auto cvt = [&](const void* src, bf16_t* dst, long n) {
    int blocks = (int)((n + 1023) / 1024); if (blocks > 4096) blocks = 4096;
    cvt_bf16_kernel<<<blocks, 256, 0, stream>>>((const float*)src, dst, n);
  };

  // 1) bf16 conversions of weights + activations feeding GEMMs
  cvt(d_in[4],  wLinear, (long)Dt * D);
  cvt(d_in[12], wGeglu,  (long)D * 2 * FF);
  cvt(d_in[14], wFfout,  (long)FF * D);
  cvt(d_in[16], wQ,      (long)D * IC);
  cvt(d_in[17], wK,      (long)D * IC);
  cvt(d_in[18], wV,      (long)D * IC);
  cvt(d_in[19], wCo,     (long)IC * D);
  cvt(d_in[23], wAq,     (long)D * D);
  cvt(d_in[24], wAk,     (long)Dt * D);
  cvt(d_in[25], wAv,     (long)Dt * D);
  cvt(d_in[26], wAo,     (long)D * D);
  cvt(d_in[1],  encB,    (long)Bn * Ltxt * Dt);
  cvt(d_in[2],  objB,    (long)Bn * Lobj * Dt);

  // 2) x = concat(hidden, obj @ linear_w + b)
  scatter_visual_kernel<<<4096, 256, 0, stream>>>((const float*)d_in[0], xF, Nv, Nx, D, Bn);
  gemm_bf16_kernel<false, false><<<dim3(10, 1, 8), 256, 0, stream>>>(
      objB, (long)Lobj * Dt, Lobj, Dt, wLinear, nullptr, D,
      (const float*)d_in[5], nullptr,
      xF + (long)Nv * D, nullptr, (long)Nx * D, D, D, Dt);

  // 3) ln1 -> bf16
  layernorm_bf16_kernel<<<Bn * Nx, 256, 0, stream>>>(
      xF, (const float*)d_in[6], (const float*)d_in[7], lnB, D);

  // 4) q,k,v projections (no bias) -> bf16
  gemm_bf16_kernel<false, true><<<dim3(4, 66, 1), 256, 0, stream>>>(
      lnB, 0, Bn * Nx, D, wQ, nullptr, IC, nullptr, nullptr, nullptr, qB, 0, IC, IC, D);
  gemm_bf16_kernel<false, true><<<dim3(4, 66, 1), 256, 0, stream>>>(
      lnB, 0, Bn * Nx, D, wK, nullptr, IC, nullptr, nullptr, nullptr, kB, 0, IC, IC, D);
  gemm_bf16_kernel<false, true><<<dim3(4, 66, 1), 256, 0, stream>>>(
      lnB, 0, Bn * Nx, D, wV, nullptr, IC, nullptr, nullptr, nullptr, vB, 0, IC, IC, D);

  // 5) masked self-attention (only first 1024 query rows are consumed)
  flash_attn_kernel<<<dim3(Nv / 128, 8, Bn), 256, 0, stream>>>(
      qB, kB, vB, (const int*)d_in[3], ctxSelf,
      Nv, Nx, IC, IC, IC,
      (long)Nx * IC, (long)Nx * IC, (long)Nv * IC,
      (long)8 * Nx * Nx, (long)Nx * Nx, Nx, 0.125f);

  // 6) self-attn out projection -> f32 (reuses xF)
  gemm_bf16_kernel<false, false><<<dim3(10, 64, 1), 256, 0, stream>>>(
      ctxSelf, 0, Bn * Nv, IC, wCo, nullptr, D,
      (const float*)d_in[20], nullptr, xF, nullptr, 0, D, D, IC);

  // 7) hs1 = hidden + tanh(alpha_attn) * attn_out
  residual_kernel<<<4096, 256, 0, stream>>>(hs1, (const float*)d_in[0], xF,
                                            (const float*)d_in[21], (long)Bn * Nv * D);

  // 8) ln2, GEGLU (fused dual-GEMM + erf-GELU), FF out
  layernorm_bf16_kernel<<<Bn * Nv, 256, 0, stream>>>(
      hs1, (const float*)d_in[8], (const float*)d_in[9], lnB, D);
  gemm_bf16_kernel<true, true><<<dim3(40, 64, 1), 256, 0, stream>>>(
      lnB, 0, Bn * Nv, D, wGeglu, wGeglu + FF, 2 * FF,
      (const float*)d_in[13], (const float*)d_in[13] + FF,
      nullptr, ffin, 0, FF, FF, D);
  gemm_bf16_kernel<false, false><<<dim3(10, 64, 1), 256, 0, stream>>>(
      ffin, 0, Bn * Nv, FF, wFfout, nullptr, D,
      (const float*)d_in[15], nullptr, xF, nullptr, 0, D, D, FF);

  // 9) hs1 += tanh(alpha_dense) * ff_out  (in-place)
  residual_kernel<<<4096, 256, 0, stream>>>(hs1, hs1, xF,
                                            (const float*)d_in[22], (long)Bn * Nv * D);

  // 10) ln3, cross-attention q/k/v
  layernorm_bf16_kernel<<<Bn * Nv, 256, 0, stream>>>(
      hs1, (const float*)d_in[10], (const float*)d_in[11], lnB, D);
  gemm_bf16_kernel<false, true><<<dim3(10, 64, 1), 256, 0, stream>>>(
      lnB, 0, Bn * Nv, D, wAq, nullptr, D, nullptr, nullptr, nullptr, qxB, 0, D, D, D);
  gemm_bf16_kernel<false, true><<<dim3(10, 5, 1), 256, 0, stream>>>(
      encB, 0, Bn * Ltxt, Dt, wAk, nullptr, D, nullptr, nullptr, nullptr, kxB, 0, D, D, Dt);
  gemm_bf16_kernel<false, true><<<dim3(10, 5, 1), 256, 0, stream>>>(
      encB, 0, Bn * Ltxt, Dt, wAv, nullptr, D, nullptr, nullptr, nullptr, vxB, 0, D, D, Dt);

  // 11) cross attention (no mask, Nkv = 77)
  flash_attn_kernel<<<dim3(Nv / 128, 20, Bn), 256, 0, stream>>>(
      qxB, kxB, vxB, nullptr, ctx2,
      Nv, Ltxt, D, D, D,
      (long)Nv * D, (long)Ltxt * D, (long)Nv * D,
      0, 0, 0, 0.125f);

  // 12) final out projection -> d_out (f32)
  gemm_bf16_kernel<false, false><<<dim3(10, 64, 1), 256, 0, stream>>>(
      ctx2, 0, Bn * Nv, D, wAo, nullptr, D,
      (const float*)d_in[27], nullptr, (float*)d_out, nullptr, 0, D, D, D);
}
